// GATv2_9509057593577
// MI455X (gfx1250) — matrix operand order repaired and checked by reading, not compiled, not run on hardware
//
#include <hip/hip_runtime.h>

// ---------------- problem constants (match reference) ----------------
#define N_GENES 64
#define BATCH   256
#define IN_DIM  1280
#define EMBED   128
#define HEADS   4
#define N_EDGES 1024
#define ETOT    (N_EDGES + N_GENES)   // +self loops = 1088
#define HID1    512
#define HID2    128
#define GC      (N_GENES * EMBED)     // 8192

// ---------------- types ----------------
typedef __attribute__((ext_vector_type(16))) __bf16 bf16x16;
typedef __attribute__((ext_vector_type(8)))  float  floatx8;
typedef __attribute__((ext_vector_type(4)))  unsigned int uint32x4;
typedef __attribute__((ext_vector_type(4)))  int          int32x4;
typedef __attribute__((ext_vector_type(8)))  int          int32x8;

union FragU {
    uint4   u[2];
    bf16x16 v;
};

__device__ __forceinline__ unsigned short f32_to_bf16(float f) {
    unsigned int u = __float_as_uint(f);
    unsigned int r = u + 0x7FFFu + ((u >> 16) & 1u);   // round-to-nearest-even
    return (unsigned short)(r >> 16);
}

// =====================================================================
// TDM 2D tile load: [rows x 32] bf16 from row-major tensor (row stride =
// strideElems) into contiguous LDS at lds_off. Tiles are always fully
// in-bounds, so tensor_dim0/1 are loop-invariant "huge" constants and the
// only per-issue variables are the global address, LDS address and rows.
// =====================================================================
#if __has_builtin(__builtin_amdgcn_tensor_load_to_lds)
#define USE_TDM 1
#else
#define USE_TDM 0
#endif

#if USE_TDM
__device__ __forceinline__ void tdm_issue(unsigned lds_off, const unsigned short* gptr,
                                          unsigned rows, unsigned long long strideElems)
{
    unsigned long long ga = (unsigned long long)gptr;
    uint32x4 g0;
    g0[0] = 1u;                                        // count=1, user-mode
    g0[1] = lds_off;                                   // lds_addr (bytes)
    g0[2] = (unsigned)ga;                              // global_addr[31:0]
    g0[3] = (unsigned)((ga >> 32) & 0x01ffffffu)       // global_addr[56:32]
          | 0x80000000u;                               // type=2 ("image")
    int32x8 g1;
    g1[0] = 0x00010000;                                // data_size = 2 bytes
    g1[1] = (int)0xffff0000u;                          // tensor_dim0 = 0x7fffffff (lo16)
    g1[2] = (int)0xffff7fffu;                          // dim0 hi16 | tensor_dim1 lo16
    g1[3] = (int)((32u << 16) | 0x7fffu);              // dim1 hi16 | tile_dim0 = 32
    g1[4] = (int)rows;                                 // tile_dim1 = rows, tile_dim2=0
    g1[5] = (int)(unsigned)strideElems;                // dim0_stride[31:0]
    g1[6] = (int)((strideElems >> 32) & 0xffffu);      // dim0_stride[47:32]
    g1[7] = 0;
    int32x4 z4 = {};
#if __clang_major__ >= 23
    int32x8 z8 = {};
    __builtin_amdgcn_tensor_load_to_lds(g0, g1, z4, z4, z8, 0);
#else
    __builtin_amdgcn_tensor_load_to_lds(g0, g1, z4, z4, 0);
#endif
}
#endif

// ---- one K-step of WMMA work for a 32x32 wave tile ----
__device__ __forceinline__ void wmma_step(const unsigned short* la, const unsigned short* lb,
                                          int wm, int wn, int lane, floatx8 acc[2][2])
{
    FragU au[2], bu[2];
    const int koffA = (lane >> 4) * 8;
    const int koffB = (lane >> 4) * 8;
#pragma unroll
    for (int s = 0; s < 2; ++s) {
        const int r = 32 * wm + 16 * s + (lane & 15);
        au[s].u[0] = *(const uint4*)(la + r * 32 + koffA);
        au[s].u[1] = *(const uint4*)(la + r * 32 + koffA + 16);
    }
#pragma unroll
    for (int j = 0; j < 2; ++j) {
        const int n = 32 * wn + 16 * j + (lane & 15);
        bu[j].u[0] = *(const uint4*)(lb + n * 32 + koffB);
        bu[j].u[1] = *(const uint4*)(lb + n * 32 + koffB + 16);
    }
#pragma unroll
    for (int s = 0; s < 2; ++s)
#pragma unroll
        for (int j = 0; j < 2; ++j)
            acc[s][j] = __builtin_amdgcn_wmma_f32_16x16x32_bf16(
                false, au[s].v, false, bu[j].v, (short)0, acc[s][j], false, false);
}

// =====================================================================
// bf16 GEMM: C[bz] = act(A[bz] @ Bt[bz]^T + bias[bz])
//   A : [M,K] bf16 row-major,  Bt : [N,K] bf16 row-major (pre-transposed W)
//   C : [M,N] fp32 or bf16 (out_bf16)
// M%128==0, N%64==0, K%32==0. 256 threads = 8 wave32.
// Block tile 128x64, wave tile 32x32 -> 4 v_wmma per K-step.
// Double-buffered TDM pipeline: wave 0 waits tile k, barrier, issues tile
// k+1 into the other buffer, all waves compute tile k. One barrier/K-step.
// =====================================================================
#define BM 128
#define BN 64
#define BK 32

__global__ void __launch_bounds__(256)
gemm_bf16_tdm(const unsigned short* __restrict__ A, const unsigned short* __restrict__ Bt,
              const float* __restrict__ bias, void* __restrict__ Cout,
              int M, int N, int K,
              long long sA, long long sB, long long sBias, long long sC,
              float neg_slope, int out_bf16)
{
    __shared__ unsigned short lds_a[2][BM * BK];
    __shared__ unsigned short lds_b[2][BN * BK];

    const int t  = threadIdx.x;
    const long long bz = blockIdx.z;
    A    += bz * sA;
    Bt   += bz * sB;
    bias += bz * sBias;

    const int m0 = blockIdx.y * BM;
    const int n0 = blockIdx.x * BN;

    const int lane = t & 31;
    const int wv   = t >> 5;       // 0..7
    const int wm   = wv >> 1;      // 0..3  -> 32-row slab
    const int wn   = wv & 1;       // 0..1  -> 32-col slab

    floatx8 acc[2][2] = {};

    const unsigned short* Abase = A  + (long long)m0 * K;
    const unsigned short* Bbase = Bt + (long long)n0 * K;
    const int nk = K >> 5;

#if USE_TDM
    const unsigned lA0 = (unsigned)(unsigned long long)(&lds_a[0][0]);
    const unsigned lA1 = (unsigned)(unsigned long long)(&lds_a[1][0]);
    const unsigned lB0 = (unsigned)(unsigned long long)(&lds_b[0][0]);
    const unsigned lB1 = (unsigned)(unsigned long long)(&lds_b[1][0]);

    if (t == 0) {                                  // prologue: tile 0 -> buf 0
        tdm_issue(lA0, Abase, BM, (unsigned long long)K);
        tdm_issue(lB0, Bbase, BN, (unsigned long long)K);
    }
    for (int ki = 0; ki < nk; ++ki) {
        const int cur = ki & 1;
        if (t == 0) __builtin_amdgcn_s_wait_tensorcnt(0);   // tile ki landed
        __syncthreads();                                    // visible to all waves
        if (t == 0 && ki + 1 < nk) {                        // prefetch tile ki+1
            const int kb = (ki + 1) << 5;
            tdm_issue(cur ? lA0 : lA1, Abase + kb, BM, (unsigned long long)K);
            tdm_issue(cur ? lB0 : lB1, Bbase + kb, BN, (unsigned long long)K);
        }
        wmma_step(&lds_a[cur][0], &lds_b[cur][0], wm, wn, lane, acc);
        // no trailing barrier: ds reads are dscnt-complete before each wmma,
        // and the next rewrite of this buffer is gated by the next barrier.
    }
#else
    for (int ki = 0; ki < nk; ++ki) {
        const int kb = ki << 5;
        {   // A tile: 4096 el, 2x uint4 per thread
            const int row  = t >> 1;
            const int koff = (t & 1) * 16;
            *(uint4*)(&lds_a[0][row * 32 + koff]) =
                *(const uint4*)(Abase + (long long)row * K + kb + koff);
            *(uint4*)(&lds_a[0][row * 32 + koff + 8]) =
                *(const uint4*)(Abase + (long long)row * K + kb + koff + 8);
        }
        {   // B tile: 2048 el, 1x uint4 per thread
            const int row  = t >> 2;
            const int koff = (t & 3) * 8;
            *(uint4*)(&lds_b[0][row * 32 + koff]) =
                *(const uint4*)(Bbase + (long long)row * K + kb + koff);
        }
        __syncthreads();
        wmma_step(&lds_a[0][0], &lds_b[0][0], wm, wn, lane, acc);
        __syncthreads();
    }
#endif

    // epilogue: C/D layout -> VGPR r: lanes 0-15 M=r, lanes 16-31 M=r+8
    float*          Cf = (float*)Cout + bz * sC;
    unsigned short* Cb = (unsigned short*)Cout + bz * sC;
    const int colbase = n0 + 32 * wn + (lane & 15);
    const float bv0 = bias[colbase];
    const float bv1 = bias[colbase + 16];
#pragma unroll
    for (int s = 0; s < 2; ++s) {
        const int rbase = m0 + 32 * wm + 16 * s + (lane >> 4) * 8;
#pragma unroll
        for (int j = 0; j < 2; ++j) {
            const int   col = colbase + 16 * j;
            const float bv  = j ? bv1 : bv0;
#pragma unroll
            for (int r = 0; r < 8; ++r) {
                float v = acc[s][j][r] + bv;
                v = v > 0.f ? v : v * neg_slope;    // branch-free activation
                const long long o = (long long)(rbase + r) * N + col;
                if (out_bf16) Cb[o] = f32_to_bf16(v);
                else          Cf[o] = v;
            }
        }
    }
}

// ---------------- fp32 -> bf16 flat convert ----------------
__global__ void convert_bf16_kernel(const float* __restrict__ in,
                                    unsigned short* __restrict__ out, long long count)
{
    long long i = (long long)blockIdx.x * blockDim.x + threadIdx.x;
    if (i < count) out[i] = f32_to_bf16(in[i]);
}

// ---------------- fp32 [K,N] -> bf16 [N,K] (batched) ----------------
__global__ void transpose_bf16_kernel(const float* __restrict__ in,
                                      unsigned short* __restrict__ out, int K, int N)
{
    const long long base = (long long)blockIdx.z * K * N;
    const int idx = blockIdx.x * blockDim.x + threadIdx.x;
    if (idx >= K * N) return;
    const int k = idx / N, n = idx % N;
    out[base + (long long)n * K + k] = f32_to_bf16(in[base + idx]);
}

// =====================================================================
// GATv2 edge scoring: e[e,b,h] = att[h] . leaky_relu(xl[src]+xr[dst], 0.2)
// =====================================================================
__global__ void __launch_bounds__(128)
edge_score_kernel(const float* __restrict__ XL, const float* __restrict__ XR,
                  const float* __restrict__ att, const long long* __restrict__ ei,
                  float* __restrict__ esc)
{
    const int e = blockIdx.x;
    const int b = blockIdx.y;
    const int t = threadIdx.x;

    int src, dst;
    if (e < N_EDGES) { src = (int)ei[e]; dst = (int)ei[N_EDGES + e]; }
    else             { src = dst = e - N_EDGES; }

    __shared__ float red[HEADS][EMBED];
#pragma unroll
    for (int h = 0; h < HEADS; ++h) {
        float a = XL[(((long long)src * BATCH + b) * HEADS + h) * EMBED + t]
                + XR[(((long long)dst * BATCH + b) * HEADS + h) * EMBED + t];
        a = a > 0.f ? a : 0.2f * a;
        red[h][t] = a * att[h * EMBED + t];
    }
    __syncthreads();
    for (int s = 64; s > 0; s >>= 1) {
        if (t < s) {
#pragma unroll
            for (int h = 0; h < HEADS; ++h) red[h][t] += red[h][t + s];
        }
        __syncthreads();
    }
    if (t < HEADS)
        esc[((long long)e * BATCH + b) * HEADS + t] = red[t][0];
}

// segment max + sum(exp): one thread per (g,b,h)
__global__ void seg_stats_kernel(const float* __restrict__ esc,
                                 const long long* __restrict__ ei,
                                 float* __restrict__ emax, float* __restrict__ den)
{
    const int idx = blockIdx.x * blockDim.x + threadIdx.x;
    if (idx >= N_GENES * BATCH * HEADS) return;
    const int h = idx & 3;
    const int b = (idx >> 2) & (BATCH - 1);
    const int g = idx >> 2 >> 8;

    float m = -3.4e38f;
    for (int e = 0; e < ETOT; ++e) {
        const int dst = (e < N_EDGES) ? (int)ei[N_EDGES + e] : e - N_EDGES;
        if (dst == g) {
            float v = esc[((long long)e * BATCH + b) * HEADS + h];
            m = v > m ? v : m;
        }
    }
    float s = 0.f;
    for (int e = 0; e < ETOT; ++e) {
        const int dst = (e < N_EDGES) ? (int)ei[N_EDGES + e] : e - N_EDGES;
        if (dst == g)
            s += __expf(esc[((long long)e * BATCH + b) * HEADS + h] - m);
    }
    emax[idx] = m;
    den[idx]  = s;
}

// alpha[e,b,h] = exp(e - emax[dst]) / den[dst]   (in place over esc)
__global__ void alpha_kernel(float* __restrict__ esc,
                             const long long* __restrict__ ei,
                             const float* __restrict__ emax,
                             const float* __restrict__ den)
{
    const int idx = blockIdx.x * blockDim.x + threadIdx.x;
    if (idx >= ETOT * BATCH * HEADS) return;
    const int h = idx & 3;
    const int b = (idx >> 2) & (BATCH - 1);
    const int e = idx >> 2 >> 8;
    const int dst = (e < N_EDGES) ? (int)ei[N_EDGES + e] : e - N_EDGES;
    const long long gi = ((long long)dst * BATCH + b) * HEADS + h;
    esc[idx] = __expf(esc[idx] - emax[gi]) / den[gi];
}

// aggregation + head-mean + bias, emitted bf16 directly in [B, G*C] layout
__global__ void __launch_bounds__(128)
aggregate_kernel(const float* __restrict__ XL, const float* __restrict__ alpha,
                 const long long* __restrict__ ei, const float* __restrict__ gbias,
                 unsigned short* __restrict__ AGGB)
{
    const int g = blockIdx.x;
    const int b = blockIdx.y;
    const int t = threadIdx.x;

    float a0 = 0.f, a1 = 0.f, a2 = 0.f, a3 = 0.f;
    for (int e = 0; e < ETOT; ++e) {
        const int dst = (e < N_EDGES) ? (int)ei[N_EDGES + e] : e - N_EDGES;
        if (dst != g) continue;
        const int src = (e < N_EDGES) ? (int)ei[e] : dst;
        const long long ab = ((long long)e * BATCH + b) * HEADS;
        const long long xb = (((long long)src * BATCH + b) * HEADS) * EMBED + t;
        a0 += alpha[ab + 0] * XL[xb];
        a1 += alpha[ab + 1] * XL[xb + EMBED];
        a2 += alpha[ab + 2] * XL[xb + 2 * EMBED];
        a3 += alpha[ab + 3] * XL[xb + 3 * EMBED];
    }
    const float v = 0.25f * (a0 + a1 + a2 + a3) + gbias[t];
    AGGB[(long long)b * GC + g * EMBED + t] = f32_to_bf16(v);
}

// final layer: out[b] = h2[b,:] . W3 + b3
__global__ void __launch_bounds__(128)
final_dot_kernel(const float* __restrict__ H2v, const float* __restrict__ W3,
                 const float* __restrict__ b3, float* __restrict__ out)
{
    const int b = blockIdx.x;
    const int t = threadIdx.x;
    __shared__ float red[HID2];
    red[t] = H2v[b * HID2 + t] * W3[t];
    __syncthreads();
    for (int s = 64; s > 0; s >>= 1) {
        if (t < s) red[t] += red[t + s];
        __syncthreads();
    }
    if (t == 0) out[b] = red[0] + b3[0];
}

// =====================================================================
extern "C" void kernel_launch(void* const* d_in, const int* in_sizes, int n_in,
                              void* d_out, int out_size, void* d_ws, size_t ws_size,
                              hipStream_t stream)
{
    const float*     pe   = (const float*)d_in[0];
    const long long* ei   = (const long long*)d_in[1];   // int64 per reference
    const float*     W_in = (const float*)d_in[2];
    const float*     b_in = (const float*)d_in[3];
    const float*     W_l  = (const float*)d_in[4];
    const float*     b_l  = (const float*)d_in[5];
    const float*     W_r  = (const float*)d_in[6];
    const float*     b_r  = (const float*)d_in[7];
    const float*     att  = (const float*)d_in[8];
    const float*     bias = (const float*)d_in[9];
    const float*     W1   = (const float*)d_in[10];
    const float*     b1   = (const float*)d_in[11];
    const float*     W2   = (const float*)d_in[12];
    const float*     b2   = (const float*)d_in[13];
    const float*     W3   = (const float*)d_in[14];
    const float*     b3   = (const float*)d_in[15];
    float*           out  = (float*)d_out;

    // ---- workspace carve-up ----
    char* w = (char*)d_ws;
    unsigned short* peB  = (unsigned short*)w; w += (long long)N_GENES * BATCH * IN_DIM * 2;
    unsigned short* WinT = (unsigned short*)w; w += (long long)N_GENES * IN_DIM * EMBED * 2;
    unsigned short* XB   = (unsigned short*)w; w += (long long)N_GENES * BATCH * EMBED * 2;
    unsigned short* WlT  = (unsigned short*)w; w += (long long)EMBED * HEADS * EMBED * 2;
    unsigned short* WrT  = (unsigned short*)w; w += (long long)EMBED * HEADS * EMBED * 2;
    unsigned short* W1T  = (unsigned short*)w; w += (long long)GC * HID1 * 2;
    unsigned short* W2T  = (unsigned short*)w; w += (long long)HID1 * HID2 * 2;
    unsigned short* AGGB = (unsigned short*)w; w += (long long)BATCH * GC * 2;
    unsigned short* H1B  = (unsigned short*)w; w += (long long)BATCH * HID1 * 2;
    float* XLp  = (float*)w; w += (long long)N_GENES * BATCH * HEADS * EMBED * 4;
    float* XRp  = (float*)w; w += (long long)N_GENES * BATCH * HEADS * EMBED * 4;
    float* ESC  = (float*)w; w += (long long)ETOT * BATCH * HEADS * 4;
    float* EMAX = (float*)w; w += (long long)N_GENES * BATCH * HEADS * 4;
    float* DEN  = (float*)w; w += (long long)N_GENES * BATCH * HEADS * 4;
    float* H2v  = (float*)w; w += (long long)BATCH * HID2 * 4;

    // ---- stage 0: one-shot bf16 conversions / weight transposes ----
    {
        long long cnt = (long long)N_GENES * BATCH * IN_DIM;
        convert_bf16_kernel<<<(int)((cnt + 255) / 256), 256, 0, stream>>>(pe, peB, cnt);
    }
    transpose_bf16_kernel<<<dim3((IN_DIM * EMBED + 255) / 256, 1, N_GENES), 256, 0, stream>>>(
        W_in, WinT, IN_DIM, EMBED);
    transpose_bf16_kernel<<<dim3((EMBED * HEADS * EMBED + 255) / 256, 1, 1), 256, 0, stream>>>(
        W_l, WlT, EMBED, HEADS * EMBED);
    transpose_bf16_kernel<<<dim3((EMBED * HEADS * EMBED + 255) / 256, 1, 1), 256, 0, stream>>>(
        W_r, WrT, EMBED, HEADS * EMBED);
    transpose_bf16_kernel<<<dim3((GC * HID1 + 255) / 256, 1, 1), 256, 0, stream>>>(
        W1, W1T, GC, HID1);
    transpose_bf16_kernel<<<dim3((HID1 * HID2 + 255) / 256, 1, 1), 256, 0, stream>>>(
        W2, W2T, HID1, HID2);

    // 1) per-gene input linear + leaky(0.01): XB[g] = pe[g] @ W_in[g] + b_in[g]
    gemm_bf16_tdm<<<dim3(EMBED / BN, BATCH / BM, N_GENES), 256, 0, stream>>>(
        peB, WinT, b_in, XB, BATCH, EMBED, IN_DIM,
        (long long)BATCH * IN_DIM, (long long)IN_DIM * EMBED, EMBED,
        (long long)BATCH * EMBED, 0.01f, /*bf16 out*/1);

    // 2) GAT source/target transforms over [G*B, C] rows (fp32 out for VALU stage)
    gemm_bf16_tdm<<<dim3((HEADS * EMBED) / BN, (N_GENES * BATCH) / BM, 1), 256, 0, stream>>>(
        XB, WlT, b_l, XLp, N_GENES * BATCH, HEADS * EMBED, EMBED, 0, 0, 0, 0, 1.0f, 0);
    gemm_bf16_tdm<<<dim3((HEADS * EMBED) / BN, (N_GENES * BATCH) / BM, 1), 256, 0, stream>>>(
        XB, WrT, b_r, XRp, N_GENES * BATCH, HEADS * EMBED, EMBED, 0, 0, 0, 0, 1.0f, 0);

    // 3) edge scores, segment softmax, aggregation
    edge_score_kernel<<<dim3(ETOT, BATCH), 128, 0, stream>>>(XLp, XRp, att, ei, ESC);
    seg_stats_kernel<<<(N_GENES * BATCH * HEADS + 255) / 256, 256, 0, stream>>>(ESC, ei, EMAX, DEN);
    alpha_kernel<<<(ETOT * BATCH * HEADS + 255) / 256, 256, 0, stream>>>(ESC, ei, EMAX, DEN);
    aggregate_kernel<<<dim3(N_GENES, BATCH), 128, 0, stream>>>(XLp, ESC, ei, bias, AGGB);

    // 4) output MLP (relu via neg_slope = 0)
    gemm_bf16_tdm<<<dim3(HID1 / BN, BATCH / BM, 1), 256, 0, stream>>>(
        AGGB, W1T, b1, H1B, BATCH, HID1, GC, 0, 0, 0, 0, 0.0f, 1);
    gemm_bf16_tdm<<<dim3(HID2 / BN, BATCH / BM, 1), 256, 0, stream>>>(
        H1B, W2T, b2, H2v, BATCH, HID2, HID1, 0, 0, 0, 0, 0.0f, 0);
    final_dot_kernel<<<BATCH, 128, 0, stream>>>(H2v, W3, b3, out);
}
